// Attention_49177375539629
// MI455X (gfx1250) — compile-verified, matrix operand order checked by reading, not run on hardware
//
#include <hip/hip_runtime.h>

typedef __bf16 bf16;
typedef __attribute__((ext_vector_type(16))) __bf16 v16bf;
typedef __attribute__((ext_vector_type(8)))  __bf16 v8bf;
typedef __attribute__((ext_vector_type(8)))  float  v8f;
typedef __attribute__((ext_vector_type(4)))  unsigned int u32x4;
typedef __attribute__((ext_vector_type(8)))  int i32x8;
typedef __attribute__((ext_vector_type(4)))  int i32x4;

union AF { v16bf v; v8bf h[2]; };

#define WMMA_BF16(a, b, c) \
  __builtin_amdgcn_wmma_f32_16x16x32_bf16(false, (a), false, (b), (short)0, (c), false, false)

// Pack two fp32 -> two bf16 in one dword (lowers to v_cvt_pk_bf16_f32).
__device__ __forceinline__ unsigned pack2_bf16(float lo, float hi) {
  unsigned short l = __builtin_bit_cast(unsigned short, (bf16)lo);
  unsigned short h = __builtin_bit_cast(unsigned short, (bf16)hi);
  return (unsigned)l | ((unsigned)h << 16);
}
__device__ __forceinline__ unsigned pack2_raw(bf16 lo, bf16 hi) {
  unsigned short l = __builtin_bit_cast(unsigned short, lo);
  unsigned short h = __builtin_bit_cast(unsigned short, hi);
  return (unsigned)l | ((unsigned)h << 16);
}

// Tensor Data Mover availability (device pass only; host pass gets fallback).
#if defined(__gfx1250__) && __has_builtin(__builtin_amdgcn_tensor_load_to_lds) && \
    __has_builtin(__builtin_amdgcn_s_wait_tensorcnt)
#define HAVE_TDM 1
#else
#define HAVE_TDM 0
#endif

#if HAVE_TDM
// 2-D TDM tile load: global (row-major, stride0 elems of 2 bytes) -> LDS (packed).
// D# packing per CDNA5 ISA ch.8: group0 = {count, lds_addr, global_addr, type=2},
// group1 = {data_size=2B, tensor_dim0/1, tile_dim0/1, tensor_dim0_stride}.
__device__ __forceinline__ void tdm_load_2d(unsigned ldsOff, const void* gptr,
                                            unsigned tile_d0, unsigned tile_d1,
                                            unsigned tensor_d0, unsigned tensor_d1,
                                            unsigned stride0) {
  unsigned long long ga = (unsigned long long)gptr;
  u32x4 g0;
  g0[0] = 1u;                                                // count=1, no gather
  g0[1] = ldsOff;                                            // lds_addr
  g0[2] = (unsigned)ga;                                      // global_addr[31:0]
  g0[3] = ((unsigned)(ga >> 32) & 0x01FFFFFFu) | 0x80000000u;// addr[56:32] | type=2
  i32x8 g1;
  g1[0] = (int)(1u << 16);                                   // data_size = 2 bytes
  g1[1] = (int)((tensor_d0 & 0xFFFFu) << 16);                // tensor_dim0[15:0]
  g1[2] = (int)(((tensor_d0 >> 16) & 0xFFFFu) | ((tensor_d1 & 0xFFFFu) << 16));
  g1[3] = (int)(((tensor_d1 >> 16) & 0xFFFFu) | ((tile_d0 & 0xFFFFu) << 16));
  g1[4] = (int)(tile_d1 & 0xFFFFu);                          // tile_dim1 (tile_dim2=0)
  g1[5] = (int)stride0;                                      // dim0_stride[31:0]
  g1[6] = 0;
  g1[7] = 0;
  i32x4 z4 = {0, 0, 0, 0};
#if __clang_major__ >= 23
  i32x8 z8 = {0, 0, 0, 0, 0, 0, 0, 0};
  __builtin_amdgcn_tensor_load_to_lds(g0, g1, z4, z4, z8, 0);
#else
  __builtin_amdgcn_tensor_load_to_lds(g0, g1, z4, z4, 0);
#endif
}
#endif

// Problem sizes: B=4, N=2048, C=1024, H=16, D=64, SCALE = 1/8.

// ---------------------------------------------------------------------------
// Kernel 1: QKV GEMM.  X[8192,1024] fp32 @ Wqkv[1024,3072] fp32
//           -> Q/K/V bf16, each [B,H,N,D].
// Block tile 128(M) x 128(N), K-step 64. 8 waves (4x2), wave tile 32x64.
// 16 WMMA per barrier pair.  Staging stores packed to b32/b64.
// ---------------------------------------------------------------------------
__global__ __launch_bounds__(256)
void qkv_gemm_kernel(const float* __restrict__ X, const float* __restrict__ W,
                     bf16* __restrict__ Qo, bf16* __restrict__ Ko, bf16* __restrict__ Vo) {
  __shared__ __align__(16) bf16 ldsA[128 * 64];   // [m][k]
  __shared__ __align__(16) bf16 ldsB[128 * 64];   // [n][k] (transposed weights)

  const int tid  = threadIdx.x;
  const int lane = tid & 31, lh = lane & 15, hi = lane >> 4;
  const int wave = tid >> 5, waveM = wave >> 1, waveN = wave & 1;
  const int rowbase = blockIdx.y * 128;
  const int colbase = blockIdx.x * 128;

  v8f acc[2][4] = {};

  const int am  = tid >> 1, ak = (tid & 1) * 32;  // A staging: 32 fp32/thread
  const int bk2 = (tid >> 3) * 2;                 // B staging: 2 k-rows x 16 n/thread
  const int bn0 = (tid & 7) * 16;

  for (int kb = 0; kb < 1024; kb += 64) {
    __syncthreads();
    {   // A tile: fp32 -> bf16, packed b64 stores
      const float4* g = (const float4*)(X + (size_t)(rowbase + am) * 1024 + kb + ak);
      #pragma unroll
      for (int j = 0; j < 8; ++j) {
        float4 f = g[j];
        uint2 p = make_uint2(pack2_bf16(f.x, f.y), pack2_bf16(f.z, f.w));
        *(uint2*)&ldsA[am * 64 + ak + j * 4] = p;
      }
    }
    {   // B tile: two k-rows -> packed (k,k+1) b32 stores into ldsB[n][k]
      const float4* g0 = (const float4*)(W + (size_t)(kb + bk2)     * 3072 + colbase + bn0);
      const float4* g1 = (const float4*)(W + (size_t)(kb + bk2 + 1) * 3072 + colbase + bn0);
      #pragma unroll
      for (int j = 0; j < 4; ++j) {
        float4 a = g0[j], b = g1[j];
        *(unsigned*)&ldsB[(bn0 + j * 4 + 0) * 64 + bk2] = pack2_bf16(a.x, b.x);
        *(unsigned*)&ldsB[(bn0 + j * 4 + 1) * 64 + bk2] = pack2_bf16(a.y, b.y);
        *(unsigned*)&ldsB[(bn0 + j * 4 + 2) * 64 + bk2] = pack2_bf16(a.z, b.z);
        *(unsigned*)&ldsB[(bn0 + j * 4 + 3) * 64 + bk2] = pack2_bf16(a.w, b.w);
      }
    }
    __syncthreads();

    #pragma unroll
    for (int kc = 0; kc < 2; ++kc) {
      AF a[2], b[4];
      #pragma unroll
      for (int ti = 0; ti < 2; ++ti) {
        const bf16* p = &ldsA[(waveM * 32 + ti * 16 + lh) * 64 + kc * 32];
        a[ti].h[0] = *(const v8bf*)(p + hi * 8);
        a[ti].h[1] = *(const v8bf*)(p + 16 + hi * 8);
      }
      #pragma unroll
      for (int tj = 0; tj < 4; ++tj) {
        const bf16* p = &ldsB[(waveN * 64 + tj * 16 + lh) * 64 + kc * 32];
        b[tj].h[0] = *(const v8bf*)(p + hi * 8);
        b[tj].h[1] = *(const v8bf*)(p + 16 + hi * 8);
      }
      #pragma unroll
      for (int ti = 0; ti < 2; ++ti)
        #pragma unroll
        for (int tj = 0; tj < 4; ++tj)
          acc[ti][tj] = WMMA_BF16(a[ti].v, b[tj].v, acc[ti][tj]);
    }
  }

  // Scatter into Q/K/V [b][h][n][d]: col -> (sel, h, d), row -> (b, n).
  #pragma unroll
  for (int ti = 0; ti < 2; ++ti) {
    #pragma unroll
    for (int tj = 0; tj < 4; ++tj) {
      const int col = colbase + waveN * 64 + tj * 16 + lh;
      const int sel = col >> 10, cc = col & 1023;
      const int h = cc >> 6, d = cc & 63;
      bf16* dst = (sel == 0) ? Qo : ((sel == 1) ? Ko : Vo);
      #pragma unroll
      for (int r = 0; r < 8; ++r) {
        const int row = rowbase + waveM * 32 + ti * 16 + r + hi * 8;
        const int bi = row >> 11, n = row & 2047;
        dst[(((size_t)bi * 16 + h) * 2048 + n) * 64 + d] = (bf16)acc[ti][tj][r];
      }
    }
  }
}

// ---------------------------------------------------------------------------
// Kernel 2: Flash attention.  Per (b,h): Q,K,V [2048,64] bf16.
// Block = 8 waves, 128 Q rows; each wave owns 16 Q rows.
// K tile staged by TDM (double-buffered, wave 0 issues stage i+1 while stage i
// computes); V tile transposed with packed b32 stores; S in registers; online
// softmax; P -> LDS -> A-fragment; O accumulated in registers.
// Output written [B,N,C] bf16 (col = h*64+d) for the proj GEMM.
// ---------------------------------------------------------------------------
__global__ __launch_bounds__(256)
void attn_kernel(const bf16* __restrict__ Q, const bf16* __restrict__ K,
                 const bf16* __restrict__ V, bf16* __restrict__ O) {
  __shared__ __align__(16) bf16 ldsK [2 * 64 * 64];   // [buf][key][d]
  __shared__ __align__(16) bf16 ldsVt[2 * 64 * 64];   // [buf][d][key]
  __shared__ __align__(16) bf16 ldsP [8 * 16 * 64];   // per-wave [row][key]

  const int tid  = threadIdx.x;
  const int lane = tid & 31, lh = lane & 15, hi = lane >> 4;
  const int wave = tid >> 5;
  const int bh = blockIdx.y;            // 0..63
  const int b  = bh >> 4, h = bh & 15;
  const size_t bhoff = (size_t)bh * (2048 * 64);
  const int qrow0 = blockIdx.x * 128 + wave * 16;

  // Q fragments (A operand, 2 K-chunks of 32 over D=64), loaded once.
  AF qa[2];
  {
    const bf16* qp = Q + bhoff + (size_t)(qrow0 + lh) * 64;
    #pragma unroll
    for (int kc = 0; kc < 2; ++kc) {
      qa[kc].h[0] = *(const v8bf*)(qp + kc * 32 + hi * 8);
      qa[kc].h[1] = *(const v8bf*)(qp + kc * 32 + 16 + hi * 8);
    }
  }

  float mrun[8], lrun[8];
  #pragma unroll
  for (int r = 0; r < 8; ++r) { mrun[r] = -1e30f; lrun[r] = 0.f; }
  v8f o[4] = {};

  const int cm2 = (tid >> 3) * 2;       // V staging: two key rows per thread
  const int ck0 = (tid & 7) * 8;        // d offset 0..56

#if HAVE_TDM
  const unsigned kLdsBase = (unsigned)(unsigned long long)(const void*)&ldsK[0];
  if (wave == 0)   // pre-issue K tile for stage 0 into buffer 0
    tdm_load_2d(kLdsBase, K + bhoff, /*tile*/64, 64, /*tensor*/64, 2048, /*stride*/64);
#endif

  for (int kb = 0, it = 0; kb < 2048; kb += 64, ++it) {
    const int cur = it & 1, nxt = cur ^ 1;
    bf16* kcur = &ldsK [cur * 4096];
    bf16* vcur = &ldsVt[cur * 4096];

    {   // stage V tile transposed: pack (key, key+1) pairs -> b32 stores
      const bf16* vg0 = V + bhoff + (size_t)(kb + cm2)     * 64 + ck0;
      const bf16* vg1 = V + bhoff + (size_t)(kb + cm2 + 1) * 64 + ck0;
      v8bf r0 = *(const v8bf*)vg0, r1 = *(const v8bf*)vg1;
      #pragma unroll
      for (int j = 0; j < 8; ++j)
        *(unsigned*)&vcur[(ck0 + j) * 64 + cm2] = pack2_raw(r0[j], r1[j]);
      if (kb + 64 < 2048)
        __builtin_prefetch(V + bhoff + (size_t)(kb + 64 + cm2) * 64 + ck0, 0, 0);
    }

#if HAVE_TDM
    if (wave == 0) {
      if (kb + 64 < 2048) {   // issue next K tile, then wait for current
        tdm_load_2d(kLdsBase + (unsigned)nxt * 8192,
                    K + bhoff + (size_t)(kb + 64) * 64, 64, 64, 64, 2048, 64);
        __builtin_amdgcn_s_wait_tensorcnt(1);
      } else {
        __builtin_amdgcn_s_wait_tensorcnt(0);
      }
    }
#else
    {   // fallback: synchronous K staging (two rows per thread)
      const bf16* kg0 = K + bhoff + (size_t)(kb + cm2)     * 64 + ck0;
      const bf16* kg1 = K + bhoff + (size_t)(kb + cm2 + 1) * 64 + ck0;
      *(v8bf*)&kcur[(cm2)     * 64 + ck0] = *(const v8bf*)kg0;
      *(v8bf*)&kcur[(cm2 + 1) * 64 + ck0] = *(const v8bf*)kg1;
    }
#endif
    __syncthreads();   // K_i + V_i visible to all waves

    // S = Q * K^T for 4 key sub-tiles of 16.
    v8f s[4] = {};
    #pragma unroll
    for (int kc = 0; kc < 2; ++kc) {
      #pragma unroll
      for (int kt = 0; kt < 4; ++kt) {
        AF bf;
        const bf16* p = &kcur[(kt * 16 + lh) * 64 + kc * 32];
        bf.h[0] = *(const v8bf*)(p + hi * 8);
        bf.h[1] = *(const v8bf*)(p + 16 + hi * 8);
        s[kt] = WMMA_BF16(qa[kc].v, bf.v, s[kt]);
      }
    }
    #pragma unroll
    for (int kt = 0; kt < 4; ++kt) s[kt] = s[kt] * 0.125f;   // SCALE = D^-0.5

    // Online softmax. Rows live in lane halves; reduce within 16-lane groups.
    #pragma unroll
    for (int r = 0; r < 8; ++r) {
      float mx = fmaxf(fmaxf(s[0][r], s[1][r]), fmaxf(s[2][r], s[3][r]));
      mx = fmaxf(mx, __shfl_xor(mx, 1, 32));
      mx = fmaxf(mx, __shfl_xor(mx, 2, 32));
      mx = fmaxf(mx, __shfl_xor(mx, 4, 32));
      mx = fmaxf(mx, __shfl_xor(mx, 8, 32));
      const float mn = fmaxf(mrun[r], mx);
      const float alpha = __expf(mrun[r] - mn);
      mrun[r] = mn;
      float rs = 0.f;
      #pragma unroll
      for (int kt = 0; kt < 4; ++kt) {
        float p = __expf(s[kt][r] - mn);
        s[kt][r] = p;
        rs += p;
      }
      rs += __shfl_xor(rs, 1, 32);
      rs += __shfl_xor(rs, 2, 32);
      rs += __shfl_xor(rs, 4, 32);
      rs += __shfl_xor(rs, 8, 32);
      lrun[r] = lrun[r] * alpha + rs;
      #pragma unroll
      for (int dt = 0; dt < 4; ++dt) o[dt][r] *= alpha;
    }

    // P: accumulator (D) layout -> LDS -> A-operand layout (same wave, in order).
    bf16* pw = &ldsP[wave * (16 * 64)];
    #pragma unroll
    for (int kt = 0; kt < 4; ++kt)
      #pragma unroll
      for (int r = 0; r < 8; ++r)
        pw[(r + hi * 8) * 64 + kt * 16 + lh] = (bf16)s[kt][r];

    #pragma unroll
    for (int kc = 0; kc < 2; ++kc) {
      AF pa;
      const bf16* pp = &pw[lh * 64 + kc * 32];
      pa.h[0] = *(const v8bf*)(pp + hi * 8);
      pa.h[1] = *(const v8bf*)(pp + 16 + hi * 8);
      #pragma unroll
      for (int dt = 0; dt < 4; ++dt) {
        AF vb;
        const bf16* vp = &vcur[(dt * 16 + lh) * 64 + kc * 32];
        vb.h[0] = *(const v8bf*)(vp + hi * 8);
        vb.h[1] = *(const v8bf*)(vp + 16 + hi * 8);
        o[dt] = WMMA_BF16(pa.v, vb.v, o[dt]);
      }
    }
    __syncthreads();   // everyone done with this stage's buffers
  }

  // Normalize and store O as [b][n][h*64+d] bf16.
  #pragma unroll
  for (int dt = 0; dt < 4; ++dt)
    #pragma unroll
    for (int r = 0; r < 8; ++r) {
      const int row = qrow0 + r + hi * 8;
      const int col = h * 64 + dt * 16 + lh;
      O[((size_t)b * 2048 + row) * 1024 + col] = (bf16)(o[dt][r] / lrun[r]);
    }
}

// ---------------------------------------------------------------------------
// Kernel 3: output projection.  O[8192,1024] bf16 @ Wproj[1024,1024] fp32
//           + b_proj -> out fp32.  128x128 tile, K-step 64.
// A tile (bf16) staged via TDM when available.
// ---------------------------------------------------------------------------
__global__ __launch_bounds__(256)
void proj_gemm_kernel(const bf16* __restrict__ A, const float* __restrict__ W,
                      const float* __restrict__ bias, float* __restrict__ out) {
  __shared__ __align__(16) bf16 ldsA[128 * 64];
  __shared__ __align__(16) bf16 ldsB[128 * 64];

  const int tid  = threadIdx.x;
  const int lane = tid & 31, lh = lane & 15, hi = lane >> 4;
  const int wave = tid >> 5, waveM = wave >> 1, waveN = wave & 1;
  const int rowbase = blockIdx.y * 128;
  const int colbase = blockIdx.x * 128;

  v8f acc[2][4] = {};

  const int am  = tid >> 1, ak = (tid & 1) * 32;
  const int bk2 = (tid >> 3) * 2;
  const int bn0 = (tid & 7) * 16;

#if HAVE_TDM
  const unsigned aLdsOff = (unsigned)(unsigned long long)(const void*)&ldsA[0];
#endif

  for (int kb = 0; kb < 1024; kb += 64) {
    __syncthreads();
#if HAVE_TDM
    if (wave == 0)   // A tile [128 rows x 64 cols] bf16, row stride 1024
      tdm_load_2d(aLdsOff, A + (size_t)rowbase * 1024 + kb,
                  /*tile*/64, 128, /*tensor*/1024, 8192, /*stride*/1024);
#else
    {   // fallback: straight 128-bit copies
      const bf16* g = A + (size_t)(rowbase + am) * 1024 + kb + ak;
      #pragma unroll
      for (int j = 0; j < 4; ++j)
        *(v8bf*)&ldsA[am * 64 + ak + j * 8] = *(const v8bf*)(g + j * 8);
    }
#endif
    {   // B tile: two k-rows -> packed (k,k+1) b32 stores into ldsB[n][k]
      const float4* g0 = (const float4*)(W + (size_t)(kb + bk2)     * 1024 + colbase + bn0);
      const float4* g1 = (const float4*)(W + (size_t)(kb + bk2 + 1) * 1024 + colbase + bn0);
      #pragma unroll
      for (int j = 0; j < 4; ++j) {
        float4 a = g0[j], b = g1[j];
        *(unsigned*)&ldsB[(bn0 + j * 4 + 0) * 64 + bk2] = pack2_bf16(a.x, b.x);
        *(unsigned*)&ldsB[(bn0 + j * 4 + 1) * 64 + bk2] = pack2_bf16(a.y, b.y);
        *(unsigned*)&ldsB[(bn0 + j * 4 + 2) * 64 + bk2] = pack2_bf16(a.z, b.z);
        *(unsigned*)&ldsB[(bn0 + j * 4 + 3) * 64 + bk2] = pack2_bf16(a.w, b.w);
      }
    }
#if HAVE_TDM
    if (wave == 0) __builtin_amdgcn_s_wait_tensorcnt(0);
#endif
    __syncthreads();

    #pragma unroll
    for (int kc = 0; kc < 2; ++kc) {
      AF a[2], b[4];
      #pragma unroll
      for (int ti = 0; ti < 2; ++ti) {
        const bf16* p = &ldsA[(waveM * 32 + ti * 16 + lh) * 64 + kc * 32];
        a[ti].h[0] = *(const v8bf*)(p + hi * 8);
        a[ti].h[1] = *(const v8bf*)(p + 16 + hi * 8);
      }
      #pragma unroll
      for (int tj = 0; tj < 4; ++tj) {
        const bf16* p = &ldsB[(waveN * 64 + tj * 16 + lh) * 64 + kc * 32];
        b[tj].h[0] = *(const v8bf*)(p + hi * 8);
        b[tj].h[1] = *(const v8bf*)(p + 16 + hi * 8);
      }
      #pragma unroll
      for (int ti = 0; ti < 2; ++ti)
        #pragma unroll
        for (int tj = 0; tj < 4; ++tj)
          acc[ti][tj] = WMMA_BF16(a[ti].v, b[tj].v, acc[ti][tj]);
    }
  }

  #pragma unroll
  for (int ti = 0; ti < 2; ++ti) {
    #pragma unroll
    for (int tj = 0; tj < 4; ++tj) {
      const int col = colbase + waveN * 64 + tj * 16 + lh;
      const float bv = bias[col];
      #pragma unroll
      for (int r = 0; r < 8; ++r) {
        const int row = rowbase + waveM * 32 + ti * 16 + r + hi * 8;
        out[(size_t)row * 1024 + col] = acc[ti][tj][r] + bv;
      }
    }
  }
}

// ---------------------------------------------------------------------------
extern "C" void kernel_launch(void* const* d_in, const int* in_sizes, int n_in,
                              void* d_out, int out_size, void* d_ws, size_t ws_size,
                              hipStream_t stream) {
  const float* x      = (const float*)d_in[0];
  const float* w_qkv  = (const float*)d_in[1];
  const float* w_proj = (const float*)d_in[2];
  const float* b_proj = (const float*)d_in[3];
  float* out = (float*)d_out;

  // Workspace: Q,K,V [B,H,N,D] bf16 + O [B,N,C] bf16 = 4 x 16 MB = 64 MB.
  const size_t QKV_ELEMS = (size_t)4 * 16 * 2048 * 64;   // 8 Mi elems each
  bf16* ws = (bf16*)d_ws;
  bf16* Qw = ws;
  bf16* Kw = ws + QKV_ELEMS;
  bf16* Vw = ws + 2 * QKV_ELEMS;
  bf16* Ow = ws + 3 * QKV_ELEMS;

  qkv_gemm_kernel <<<dim3(24, 64), 256, 0, stream>>>(x, w_qkv, Qw, Kw, Vw);
  attn_kernel     <<<dim3(16, 64), 256, 0, stream>>>(Qw, Kw, Vw, Ow);
  proj_gemm_kernel<<<dim3(8, 64),  256, 0, stream>>>(Ow, w_proj, b_proj, out);
}
